// Linear3BitEfficient_79405355369202
// MI455X (gfx1250) — matrix operand order repaired and checked by reading, not compiled, not run on hardware
//
#include <hip/hip_runtime.h>

// gfx1250 WMMA types
typedef __attribute__((ext_vector_type(16))) _Float16 v16h;
typedef __attribute__((ext_vector_type(8)))  _Float16 v8h;
typedef __attribute__((ext_vector_type(4)))  _Float16 v4h;
typedef __attribute__((ext_vector_type(8)))  float    v8f;
typedef __attribute__((ext_vector_type(4)))  float    v4f;
typedef __attribute__((ext_vector_type(4)))  int      v4i;

#define TILE_M 128
#define TILE_N 128
#define TILE_K 32
#define LDS_STRIDE 40   // 32 + 8 halves of pad -> 80B row stride, spreads LDS banks

// Workgroup barrier WITHOUT the acq_rel vmem drain __syncthreads() implies:
// drains DScnt (LDS stores visible before signal) but leaves global LOADcnt
// alone -> next-tile global loads stay in flight across the barrier.
__device__ __forceinline__ void wg_barrier() {
    asm volatile("s_wait_dscnt 0x0\n\t"
                 "s_barrier_signal -1\n\t"
                 "s_barrier_wait -1"
                 ::: "memory");
}

// out[M,N] = x[M,K] * dequant(W)[N,K]^T + bias[N]
// W: 3-bit, group size 32 along K, per-group fp16 scale.
// Compute-bound (AI ~ 4000 flop/byte) -> dense f16 WMMA GEMM, on-the-fly dequant.
// 4 waves/block, 64x64 tile per wave: 16 WMMAs per wave per K-step, 4x fragment
// reuse on both operands. Ping-pong LDS: ONE barrier per K-step. Register
// double-buffered global loads span barrier + compute.
__global__ __launch_bounds__(128)
void Linear3Bit_wmma_kernel(const float* __restrict__ x,
                            const int* __restrict__ wpacked,
                            const _Float16* __restrict__ scales,
                            const _Float16* __restrict__ bias,
                            float* __restrict__ out,
                            int M, int N, int K)
{
    __shared__ __align__(32) _Float16 As[2][TILE_M * LDS_STRIDE];
    __shared__ __align__(32) _Float16 Bs[2][TILE_N * LDS_STRIDE];

    const int tid     = threadIdx.x;        // 0..127
    const int lane    = tid & 31;
    const int lane_lo = lane & 15;
    const int lane_hi = lane >> 4;          // 0 or 1
    const int wave    = tid >> 5;           // 0..3

    const int nBase = blockIdx.x * TILE_N;
    const int mBase = blockIdx.y * TILE_M;

    const int wave_m = (wave & 1) * 64;     // 0,64
    const int wave_n = (wave >> 1) * 64;    // 0,64

    const int groupsPerRow = K / 32;        // 128

    // ---- per-thread A staging map (8 x float4 per thread, coalesced) ----
    unsigned aGOff[8];
    int aOff[8];
    #pragma unroll
    for (int i = 0; i < 8; ++i) {
        int f   = tid + i * 128;            // float4 slot 0..1023
        int row = f >> 3;                   // 8 float4 per 32-wide k-row
        int kv  = (f & 7) << 2;
        aGOff[i] = (unsigned)(((mBase + row) * K + kv) * 4);   // byte offset
        aOff[i]  = row * LDS_STRIDE + kv;
    }

    // ---- per-thread B staging map: ONE thread per quant group (n-row) ----
    const int gRow = tid;                   // 0..127 : n row within tile
    const unsigned wGOff = (unsigned)((nBase + gRow) * groupsPerRow * 16 * 4);
    const unsigned sGOff = (unsigned)((nBase + gRow) * groupsPerRow * 2);
    const int bOff = gRow * LDS_STRIDE;

    v8f acc[4][4] = {};                     // 16 WMMA accumulators per wave

    // ---- double-buffer registers ----
    v4f aReg[8];
    v4i bReg[4];
    float sReg;

    auto issue_loads = [&](int kb) {
        const char* xk = (const char*)x + (size_t)kb * 4;               // uniform
        #pragma unroll
        for (int i = 0; i < 8; ++i)
            aReg[i] = *(const v4f*)(xk + aGOff[i]);
        const char* wk = (const char*)wpacked + (size_t)(kb >> 5) * 64; // uniform
        #pragma unroll
        for (int j = 0; j < 4; ++j)
            bReg[j] = *(const v4i*)(wk + wGOff + 16 * j);
        const char* sk = (const char*)scales + (size_t)(kb >> 5) * 2;   // uniform
        sReg = (float)*(const _Float16*)(sk + sGOff);
    };

    // One pipeline step: stage regs -> LDS buf, issue loads for kn, barrier,
    // then 16 WMMAs from LDS buf (next-tile global loads in flight throughout).
    auto step = [&](_Float16* Asb, _Float16* Bsb, int kn) {
        // ---- stage current tile into LDS (loadcnt waited here, at first use) ----
        #pragma unroll
        for (int i = 0; i < 8; ++i) {
            v4h h;
            h[0] = (_Float16)aReg[i][0]; h[1] = (_Float16)aReg[i][1];
            h[2] = (_Float16)aReg[i][2]; h[3] = (_Float16)aReg[i][3];
            *(v4h*)&Asb[aOff[i]] = h;
        }
        {
            float s2   = sReg * (2.0f / 7.0f);   // q*(2s/7) - s == ((q*2/7)-1)*s
            float sneg = -sReg;
            _Float16* bDst = Bsb + bOff;
            #pragma unroll
            for (int v = 0; v < 4; ++v) {        // 4 dwords -> 8 halves each
                v8h h8;
                #pragma unroll
                for (int j = 0; j < 4; ++j) {
                    int w = bReg[v][j];
                    h8[2*j]   = (_Float16)__builtin_fmaf((float)(w & 7),        s2, sneg);
                    h8[2*j+1] = (_Float16)__builtin_fmaf((float)((w >> 3) & 7), s2, sneg);
                }
                *(v8h*)(bDst + 8 * v) = h8;
            }
        }

        // ---- next tile's global loads: in flight across barrier + compute ----
        issue_loads(kn);

        wg_barrier();

        // ---- fragments per ISA VGPR layouts ----
        // A 16x32 f16: lanes 0-15 hold K 0..7 & 16..23, lanes 16-31 hold K 8..15 & 24..31
        v16h afrag[4];
        #pragma unroll
        for (int tm = 0; tm < 4; ++tm) {
            const _Float16* p = &Asb[(wave_m + tm * 16 + lane_lo) * LDS_STRIDE + lane_hi * 8];
            v8h a0 = *(const v8h*)p;
            v8h a1 = *(const v8h*)(p + 16);
            #pragma unroll
            for (int e = 0; e < 8; ++e) { afrag[tm][e] = a0[e]; afrag[tm][8 + e] = a1[e]; }
        }
        // B 32x16 f16: lane = column; lanes 0-15 hold K 0..7 & 16..23, lanes 16-31 hold K 8..15 & 24..31
        // Load B per-tn to limit live registers; 4 WMMAs consume each B fragment.
        #pragma unroll
        for (int tn = 0; tn < 4; ++tn) {
            const _Float16* p = &Bsb[(wave_n + tn * 16 + lane_lo) * LDS_STRIDE + lane_hi * 8];
            v8h b0 = *(const v8h*)p;
            v8h b1 = *(const v8h*)(p + 16);
            v16h bfrag;
            #pragma unroll
            for (int e = 0; e < 8; ++e) { bfrag[e] = b0[e]; bfrag[8 + e] = b1[e]; }
            #pragma unroll
            for (int tm = 0; tm < 4; ++tm)
                acc[tm][tn] = __builtin_amdgcn_wmma_f32_16x16x32_f16(
                    false, afrag[tm], false, bfrag,
                    (short)0, acc[tm][tn], false, false);
        }
    };

    issue_loads(0);                         // prologue: tile 0 in flight

    // Unrolled by 2 so ping-pong buffer parity is static (immediate ds offsets).
    for (int kb = 0; kb < K; kb += 2 * TILE_K) {
        int k1 = kb + TILE_K;               // always < K (K multiple of 64)
        int k2 = kb + 2 * TILE_K;
        k2 = (k2 < K) ? k2 : 0;             // harmless redundant reload on last iter
        step(As[0], Bs[0], k1);
        step(As[1], Bs[1], k2);
    }

    // ---- epilogue: C/D layout -> global, add bias ----
    // VGPR r: lanes 0-15 -> row r, lanes 16-31 -> row r+8; lane_lo -> column
    #pragma unroll
    for (int tm = 0; tm < 4; ++tm) {
        int row0 = mBase + wave_m + tm * 16 + lane_hi * 8;
        #pragma unroll
        for (int tn = 0; tn < 4; ++tn) {
            int n = nBase + wave_n + tn * 16 + lane_lo;
            float bv = (float)bias[n];
            #pragma unroll
            for (int r = 0; r < 8; ++r) {
                out[(size_t)(row0 + r) * N + n] = acc[tm][tn][r] + bv;
            }
        }
    }
}

extern "C" void kernel_launch(void* const* d_in, const int* in_sizes, int n_in,
                              void* d_out, int out_size, void* d_ws, size_t ws_size,
                              hipStream_t stream) {
    const float*    x  = (const float*)d_in[0];
    const int*      wp = (const int*)d_in[1];
    const _Float16* sc = (const _Float16*)d_in[2];
    const _Float16* bs = (const _Float16*)d_in[3];
    float* out = (float*)d_out;

    const int K = 4096;                 // IN_FEATURES
    const int N = in_sizes[3];          // OUT_FEATURES (bias length) = 4096
    const int M = in_sizes[0] / K;      // 8 * 2048 = 16384

    dim3 grid(N / TILE_N, M / TILE_M);  // 32 x 128
    Linear3Bit_wmma_kernel<<<grid, 128, 0, stream>>>(x, wp, sc, bs, out, M, N, K);
}
